// PositionAttention_60816736911963
// MI455X (gfx1250) — compile-verified, matrix operand order checked by reading, not run on hardware
//
#include <hip/hip_runtime.h>
#include <hip/hip_bf16.h>
#include <math.h>

#define BB 4
#define LL 1024
#define HDIM 512
#define NH 8
#define DH 64
#define BH (BB*NH)      // 32
#define MROWS (BB*LL)   // 4096

typedef _Float16 half_t;
typedef __attribute__((ext_vector_type(16))) _Float16 v16h;
typedef __attribute__((ext_vector_type(8)))  _Float16 v8h;
typedef __attribute__((ext_vector_type(8)))  float    v8f;
typedef __attribute__((ext_vector_type(4)))  float    v4f;

// A/B fragment (16x32 f16): lane m=lane&15 (M for A, N for B), hsel=lane>>4.
// halves 0..7  -> K = 8*hsel + 0..7      (contiguous)
// halves 8..15 -> K = 16 + 8*hsel + 0..7 (contiguous)
__device__ __forceinline__ v16h load_frag_f32(const float* __restrict__ rowp, int hsel) {
  const v4f* p0 = (const v4f*)(rowp + 8*hsel);
  const v4f* p1 = (const v4f*)(rowp + 16 + 8*hsel);
  v4f a0 = p0[0], a1 = p0[1], b0 = p1[0], b1 = p1[1];
  v16h r;
#pragma unroll
  for (int i = 0; i < 4; ++i) {
    r[i]      = (half_t)a0[i];
    r[4 + i]  = (half_t)a1[i];
    r[8 + i]  = (half_t)b0[i];
    r[12 + i] = (half_t)b1[i];
  }
  return r;
}

__device__ __forceinline__ v16h load_frag_f16(const half_t* __restrict__ rowp, int hsel) {
  v8h a = *(const v8h*)(rowp + 8*hsel);
  v8h b = *(const v8h*)(rowp + 16 + 8*hsel);
  v16h r;
#pragma unroll
  for (int i = 0; i < 8; ++i) { r[i] = a[i]; r[8 + i] = b[i]; }
  return r;
}

// ---------------------------------------------------------------------------
// Kernel 1: QKV projection.  q/k/v = X @ W.T + b, head-split, stored f16.
// Q,K stored [bh][l][d] (row-major for A/B frags of S = Q K^T).
// V stored TRANSPOSED [bh][d][l] so the O = P V B-fragment is two b128 loads;
// the transposed store is one packed v8h store per lane (8 consecutive l).
// which==0 (Q) folds the 1/sqrt(D)=0.125 score scale into the stored values.
// ---------------------------------------------------------------------------
__global__ __launch_bounds__(128) void qkv_proj_kernel(
    const float* __restrict__ query, const float* __restrict__ key_,
    const float* __restrict__ value, const float* __restrict__ w,
    const float* __restrict__ bias, half_t* __restrict__ qh,
    half_t* __restrict__ kh, half_t* __restrict__ vt)
{
  int wave  = blockIdx.x * (blockDim.x >> 5) + (threadIdx.x >> 5);
  int lane  = threadIdx.x & 31;
  int ntile = wave & 31;          // 32 col tiles (512/16)
  int mtile = (wave >> 5) & 255;  // 256 row tiles (4096/16)
  int which = wave >> 13;         // 0=q,1=k,2=v

  const float* src  = (which == 0) ? query : (which == 1) ? key_ : value;
  const float* wrow = w + (size_t)which * HDIM * HDIM;

  int m = lane & 15, hsel = lane >> 4;
  const float* arow = src  + (size_t)(mtile * 16 + m) * HDIM;
  const float* brow = wrow + (size_t)(ntile * 16 + m) * HDIM; // B lane index == n

  v8f acc = {};
  for (int kb = 0; kb < HDIM; kb += 32) {
    v16h af = load_frag_f32(arow + kb, hsel);
    v16h bf = load_frag_f32(brow + kb, hsel);
    acc = __builtin_amdgcn_wmma_f32_16x16x32_f16(false, af, false, bf,
                                                 (short)0, acc, false, false);
  }

  int col = ntile * 16 + m;       // output column (C layout: lane&15 == N)
  float bval = bias[which * HDIM + col];
  int hh = col >> 6;              // head
  int d  = col & 63;

  if (which == 2) {
    // V transposed: Vt[(bh*DH + d)*LL + l]; 8 consecutive l per lane -> 1 store
    int gm0 = mtile * 16 + 8 * hsel;     // rows gm0..gm0+7 (same batch b)
    int b   = gm0 >> 10;
    int l0  = gm0 & 1023;
    v8h pack;
#pragma unroll
    for (int r = 0; r < 8; ++r) pack[r] = (half_t)(acc[r] + bval);
    *(v8h*)(vt + ((size_t)((b * NH + hh) * DH + d)) * LL + l0) = pack;
  } else {
    float scale = (which == 0) ? 0.125f : 1.0f;
    half_t* dst = (which == 0) ? qh : kh;
#pragma unroll
    for (int r = 0; r < 8; ++r) {
      int gm = mtile * 16 + r + 8 * hsel;  // global row in [0,4096)
      int b  = gm >> 10;
      int l  = gm & 1023;
      float v = (acc[r] + bval) * scale;
      dst[((size_t)((b * NH + hh) * LL + l)) * DH + d] = (half_t)v;
    }
  }
}

// ---------------------------------------------------------------------------
// Kernel 2: flash attention. One wave per (bh, 16-row q-tile).
// Streams k in blocks of 32 columns: 4 WMMAs for S, NT bias+mask loads,
// online softmax (shfl_xor over 16-lane halves), P via LDS re-layout,
// 4 WMMAs for O against transposed V (pure b128 loads).
// Bias (128MB) + mask (32MB) are single-pass -> non-temporal so the 16MB
// f16 workspace (re-read ~64x) stays resident in the 192MB L2.
// ---------------------------------------------------------------------------
__global__ __launch_bounds__(256) void flash_attn_kernel(
    const half_t* __restrict__ qh, const half_t* __restrict__ kh,
    const half_t* __restrict__ vt, const float* __restrict__ biasT,
    const unsigned char* __restrict__ mask, half_t* __restrict__ attn_out)
{
  __shared__ __align__(16) half_t pbuf[8][16][40];  // 40-half row pad: bank-safe

  int wid  = threadIdx.x >> 5;
  int lane = threadIdx.x & 31;
  int wave = blockIdx.x * 8 + wid;
  int qt = wave & 63;             // 64 q-tiles per head
  int bh = wave >> 6;             // 0..31
  int m = lane & 15, hsel = lane >> 4;
  int qbase = qt * 16;

  const half_t* Q  = qh + (size_t)bh * LL * DH;
  const half_t* K  = kh + (size_t)bh * LL * DH;
  const half_t* Vt = vt + (size_t)bh * DH * LL;   // [d][l]

  v16h qf0 = load_frag_f16(Q + (size_t)(qbase + m) * DH,      hsel);
  v16h qf1 = load_frag_f16(Q + (size_t)(qbase + m) * DH + 32, hsel);

  v8f oacc[4] = {v8f{}, v8f{}, v8f{}, v8f{}};
  float rm[8], rl[8];
#pragma unroll
  for (int r = 0; r < 8; ++r) { rm[r] = -1e30f; rl[r] = 0.0f; }

  for (int kt = 0; kt < LL / 16; kt += 2) {
    // ---- scores: two 16x16 tiles (scale pre-folded into Q) ----
    v8f s0 = {}, s1 = {};
    {
      v16h kf;
      kf = load_frag_f16(K + (size_t)(kt * 16 + m) * DH, hsel);
      s0 = __builtin_amdgcn_wmma_f32_16x16x32_f16(false, qf0, false, kf, (short)0, s0, false, false);
      kf = load_frag_f16(K + (size_t)(kt * 16 + m) * DH + 32, hsel);
      s0 = __builtin_amdgcn_wmma_f32_16x16x32_f16(false, qf1, false, kf, (short)0, s0, false, false);
      kf = load_frag_f16(K + (size_t)((kt + 1) * 16 + m) * DH, hsel);
      s1 = __builtin_amdgcn_wmma_f32_16x16x32_f16(false, qf0, false, kf, (short)0, s1, false, false);
      kf = load_frag_f16(K + (size_t)((kt + 1) * 16 + m) * DH + 32, hsel);
      s1 = __builtin_amdgcn_wmma_f32_16x16x32_f16(false, qf1, false, kf, (short)0, s1, false, false);
    }

    // ---- bias (kernel transposed: contiguous along q) + boolean mask ----
    // Single-pass data: non-temporal loads (TH=NT).
    int k0 = kt * 16 + m, k1 = (kt + 1) * 16 + m;
    const v4f* bp0 = (const v4f*)(biasT + ((size_t)bh * LL + k0) * LL + qbase + 8 * hsel);
    const v4f* bp1 = (const v4f*)(biasT + ((size_t)bh * LL + k1) * LL + qbase + 8 * hsel);
    v4f b00 = __builtin_nontemporal_load(bp0);
    v4f b01 = __builtin_nontemporal_load(bp0 + 1);
    v4f b10 = __builtin_nontemporal_load(bp1);
    v4f b11 = __builtin_nontemporal_load(bp1 + 1);
    const unsigned char* mp = mask + ((size_t)bh * LL + qbase + 8 * hsel) * LL;
#pragma unroll
    for (int r = 0; r < 8; ++r) {
      float v0 = s0[r] + ((r < 4) ? b00[r] : b01[r - 4]);
      float v1 = s1[r] + ((r < 4) ? b10[r] : b11[r - 4]);
      unsigned char m0 = __builtin_nontemporal_load(mp + (size_t)r * LL + k0);
      unsigned char m1 = __builtin_nontemporal_load(mp + (size_t)r * LL + k1);
      if (m0) v0 = -1e30f;
      if (m1) v1 = -1e30f;
      s0[r] = v0; s1[r] = v1;
    }

    // ---- online softmax; row M=r+8*hsel lives in one 16-lane half ----
#pragma unroll
    for (int r = 0; r < 8; ++r) {
      float mx = fmaxf(s0[r], s1[r]);
#pragma unroll
      for (int off = 1; off < 16; off <<= 1)
        mx = fmaxf(mx, __shfl_xor(mx, off, 32));
      float nm   = fmaxf(rm[r], mx);
      float corr = __expf(rm[r] - nm);
      float p0   = __expf(s0[r] - nm);
      float p1   = __expf(s1[r] - nm);
      float ls   = p0 + p1;
#pragma unroll
      for (int off = 1; off < 16; off <<= 1)
        ls += __shfl_xor(ls, off, 32);
      rl[r] = rl[r] * corr + ls;
      rm[r] = nm;
      oacc[0][r] *= corr; oacc[1][r] *= corr;
      oacc[2][r] *= corr; oacc[3][r] *= corr;
      pbuf[wid][r + 8 * hsel][m]      = (half_t)p0;
      pbuf[wid][r + 8 * hsel][16 + m] = (half_t)p1;
    }

    // ---- O += P(16x32) @ V(32x64): P via LDS re-layout, V from Vt (b128) ----
    v16h pf = load_frag_f16(&pbuf[wid][m][0], hsel);
#pragma unroll
    for (int nt = 0; nt < 4; ++nt) {
      // B[k][n] = V[kt*16+k][nt*16+n] = Vt[nt*16+n][kt*16+k]: contiguous in k
      v16h vf = load_frag_f16(Vt + (size_t)(nt * 16 + m) * LL + kt * 16, hsel);
      oacc[nt] = __builtin_amdgcn_wmma_f32_16x16x32_f16(false, pf, false, vf,
                                                        (short)0, oacc[nt], false, false);
    }
  }

  // ---- normalize, merge heads, store f16 for output projection ----
  int b = bh >> 3, h = bh & 7;
#pragma unroll
  for (int r = 0; r < 8; ++r) {
    int q = qbase + r + 8 * hsel;
    float inv = 1.0f / rl[r];
    size_t base = ((size_t)(b * LL + q)) * HDIM + h * DH;
#pragma unroll
    for (int nt = 0; nt < 4; ++nt)
      attn_out[base + nt * 16 + m] = (half_t)(oacc[nt][r] * inv);
  }
}

// ---------------------------------------------------------------------------
// Kernel 3: output projection. out = attn @ out_w.T + out_b (f32 result).
// ---------------------------------------------------------------------------
__global__ __launch_bounds__(128) void out_proj_kernel(
    const half_t* __restrict__ attn, const float* __restrict__ ow,
    const float* __restrict__ ob, float* __restrict__ out)
{
  int wave  = blockIdx.x * (blockDim.x >> 5) + (threadIdx.x >> 5);
  int lane  = threadIdx.x & 31;
  int ntile = wave & 31;
  int mtile = wave >> 5;
  int m = lane & 15, hsel = lane >> 4;

  const half_t* arow = attn + (size_t)(mtile * 16 + m) * HDIM;
  const float*  brow = ow   + (size_t)(ntile * 16 + m) * HDIM;

  v8f acc = {};
  for (int kb = 0; kb < HDIM; kb += 32) {
    v16h af = load_frag_f16(arow + kb, hsel);
    v16h bf = load_frag_f32(brow + kb, hsel);
    acc = __builtin_amdgcn_wmma_f32_16x16x32_f16(false, af, false, bf,
                                                 (short)0, acc, false, false);
  }

  int col  = ntile * 16 + m;
  float bv = ob[col];
#pragma unroll
  for (int r = 0; r < 8; ++r) {
    int gm = mtile * 16 + r + 8 * hsel;
    out[(size_t)gm * HDIM + col] = acc[r] + bv;
  }
}

extern "C" void kernel_launch(void* const* d_in, const int* in_sizes, int n_in,
                              void* d_out, int out_size, void* d_ws, size_t ws_size,
                              hipStream_t stream) {
  const float*         query     = (const float*)d_in[0];
  const float*         key_      = (const float*)d_in[1];
  const float*         value     = (const float*)d_in[2];
  const unsigned char* attn_mask = (const unsigned char*)d_in[3];  // bool
  // d_in[4] = key_padding_mask scalar (unused)
  const float*         kernelT   = (const float*)d_in[5];
  const float*         ipw       = (const float*)d_in[6];
  const float*         ipb       = (const float*)d_in[7];
  const float*         ow        = (const float*)d_in[8];
  const float*         obias     = (const float*)d_in[9];

  half_t* qh = (half_t*)d_ws;
  half_t* kh = qh + (size_t)BH * LL * DH;
  half_t* vt = kh + (size_t)BH * LL * DH;   // transposed V [bh][d][l]
  half_t* at = vt + (size_t)BH * DH * LL;   // total 16 MB of f16 scratch

  // 3 * 256 * 32 tiles, 4 waves per 128-thread block
  qkv_proj_kernel<<<(3 * 256 * 32) / 4, 128, 0, stream>>>(
      query, key_, value, ipw, ipb, qh, kh, vt);

  // 32 heads * 64 q-tiles, 8 waves per 256-thread block
  flash_attn_kernel<<<(BH * 64) / 8, 256, 0, stream>>>(
      qh, kh, vt, kernelT, attn_mask, at);

  out_proj_kernel<<<(256 * 32) / 4, 128, 0, stream>>>(at, ow, obias, (float*)d_out);
}
